// TreeLSTM_46797963657349
// MI455X (gfx1250) — matrix operand order repaired and checked by reading, not compiled, not run on hardware
//
#include <hip/hip_runtime.h>
#include <math.h>

// ---------------------------------------------------------------------------
// TreeLSTM on MI455X (gfx1250): bf16 WMMA GEMMs + fused elementwise kernels.
// Forest topology is compile-time constant (T=64 trees, 4-ary, 6 levels).
// GEMM: double-buffered LDS with global_load_async_to_lds_b128 (ASYNCcnt),
// block tile 64x256, wave tile 32x64 (8 wmma per k-step per wave).
// All GEMM shapes are exact multiples of the block tile.
// ---------------------------------------------------------------------------

typedef __bf16 bf16;
typedef __attribute__((ext_vector_type(16))) __bf16 bf16x16;
typedef __attribute__((ext_vector_type(8)))  __bf16 bf16x8;
typedef __attribute__((ext_vector_type(8)))  float  f32x8;
typedef __attribute__((ext_vector_type(4)))  int    i32x4;

static constexpr int T_TREES   = 64;
static constexpr int NPT       = 1365;           // nodes per tree
static constexpr int NTOT      = T_TREES * NPT;  // 87360
static constexpr int IN_FEAT   = 768;
static constexpr int HDIM      = 512;
static constexpr int H3        = 1536;
static constexpr int HS        = 1024;
static constexpr int LEAVES_PT = 1024;
static constexpr int PAR_PT    = 341;            // per-tree non-leaf nodes
static constexpr int NPAR      = T_TREES * PAR_PT; // 21824

__device__ __forceinline__ float sigf(float x) { return 1.0f / (1.0f + expf(-x)); }

// ---------------------------------------------------------------------------
// 16B global -> LDS copy: async (ASYNCcnt) when toolchain exposes it.
// ---------------------------------------------------------------------------
#if __has_builtin(__builtin_amdgcn_global_load_async_to_lds_b128)
#define ATHENA_ASYNC_LDS 1
#else
#define ATHENA_ASYNC_LDS 0
#endif

typedef __attribute__((address_space(1))) i32x4 as1_i32x4;
typedef __attribute__((address_space(3))) i32x4 as3_i32x4;

__device__ __forceinline__ void cp16(bf16* lds, const bf16* g) {
#if ATHENA_ASYNC_LDS
    __builtin_amdgcn_global_load_async_to_lds_b128(
        (as1_i32x4*)(void*)g, (as3_i32x4*)(void*)lds, 0, 0);
#else
    *(bf16x8*)lds = *(const bf16x8*)g;
#endif
}

template <int N>
__device__ __forceinline__ void wait_async() {
#if ATHENA_ASYNC_LDS
#if __has_builtin(__builtin_amdgcn_s_wait_asynccnt)
    __builtin_amdgcn_s_wait_asynccnt(N);
#else
    asm volatile("s_wait_asynccnt %0" ::"i"(N) : "memory");
#endif
#endif
}

// ---------------------------------------------------------------------------
// GEMM: C = A(MxK, bf16 row-major) * W(NxK, bf16 row-major)^T  [+bias][+resid]
// Block tile 64(M) x 256(N), K-step 32, 8 waves (2x4), wave tile 32x64.
// Requires: M % 64 == 0, N % 256 == 0, K % 32 == 0 (true for every call here).
// ---------------------------------------------------------------------------
#define TM 64
#define TN 256
#define TK 32
#define KP 8   // LDS pad -> row stride 40 elems (80B, keeps 16B chunks aligned)

__global__ __launch_bounds__(256)
void gemm_bf16_wmma(const bf16* __restrict__ A, int lda,
                    const bf16* __restrict__ W, int ldw,
                    const float* __restrict__ bias,
                    const float* __restrict__ Crsd, int ldr,
                    float* __restrict__ Cf, int ldcf,
                    bf16*  __restrict__ Cb, int ldcb,
                    int M, int N, int K)
{
    __shared__ bf16 As[2][TM][TK + KP];
    __shared__ bf16 Bs[2][TN][TK + KP];

    const int bm = blockIdx.y * TM;
    const int bn = blockIdx.x * TN;
    const int tid  = threadIdx.x;
    const int lane = tid & 31;
    const int wave = tid >> 5;
    const int wm = (wave >> 2) * 32;   // 0 or 32
    const int wn = (wave & 3) * 64;    // 0,64,128,192
    const int r  = lane & 15;
    const int hf = lane >> 4;

    // per-thread staging chunk coordinates (16B each)
    const int sr = tid >> 2;           // 0..63
    const int sc = (tid & 3) * 8;      // 0,8,16,24
    const bf16* gA = A + (long long)(bm + sr) * lda + sc;
    const bf16* gB[4];
    #pragma unroll
    for (int q = 0; q < 4; ++q)
        gB[q] = W + (long long)(bn + q * 64 + sr) * ldw + sc;

    f32x8 acc[2][4] = {};

    const int nt = K / TK;             // >= 16 for all shapes used here

    // stage tile t into buffer buf (5 async ops per thread per tile)
    auto stage = [&](int t, int buf) {
        int k0 = t * TK;
        cp16(&As[buf][sr][sc], gA + k0);
        #pragma unroll
        for (int q = 0; q < 4; ++q)
            cp16(&Bs[buf][q * 64 + sr][sc], gB[q] + k0);
    };

    union Frag { bf16x16 v; bf16x8 h[2]; };

    auto compute = [&](int cur) {
        Frag a0, a1, b[4];
        a0.h[0] = *(const bf16x8*)(&As[cur][wm + r][hf * 8]);
        a0.h[1] = *(const bf16x8*)(&As[cur][wm + r][16 + hf * 8]);
        a1.h[0] = *(const bf16x8*)(&As[cur][wm + 16 + r][hf * 8]);
        a1.h[1] = *(const bf16x8*)(&As[cur][wm + 16 + r][16 + hf * 8]);
        #pragma unroll
        for (int j = 0; j < 4; ++j) {
            b[j].h[0] = *(const bf16x8*)(&Bs[cur][wn + j * 16 + r][hf * 8]);
            b[j].h[1] = *(const bf16x8*)(&Bs[cur][wn + j * 16 + r][16 + hf * 8]);
        }
        #pragma unroll
        for (int j = 0; j < 4; ++j) {
            acc[0][j] = __builtin_amdgcn_wmma_f32_16x16x32_bf16(false, a0.v, false, b[j].v, (short)0, acc[0][j], false, false);
            acc[1][j] = __builtin_amdgcn_wmma_f32_16x16x32_bf16(false, a1.v, false, b[j].v, (short)0, acc[1][j], false, false);
        }
    };

    stage(0, 0);
    for (int t = 0; t < nt - 1; ++t) {
        const int cur = t & 1;
        stage(t + 1, cur ^ 1);
        __builtin_prefetch(gA + (t + 2) * TK, 0, 1);   // speculative; arena-backed
        wait_async<5>();       // in-order completion => tile t resident
        __syncthreads();       // all waves' copies of tile t visible
        compute(cur);
        __syncthreads();       // all waves done reading buffer `cur`
    }
    wait_async<0>();
    __syncthreads();
    compute((nt - 1) & 1);

    // Epilogue: C/D layout -> lane col = lane&15, row = vgpr + 8*(lane>>4)
    #pragma unroll
    for (int i = 0; i < 2; ++i) {
        #pragma unroll
        for (int j = 0; j < 4; ++j) {
            const int col = bn + wn + j * 16 + r;
            const float bv = bias ? bias[col] : 0.0f;
            #pragma unroll
            for (int v = 0; v < 8; ++v) {
                const int row = bm + wm + i * 16 + v + 8 * hf;
                float x = acc[i][j][v] + bv;
                if (Crsd) x += Crsd[(long long)row * ldr + col];
                if (Cf) Cf[(long long)row * ldcf + col] = x;
                if (Cb) Cb[(long long)row * ldcb + col] = (bf16)x;
            }
        }
    }
}

// ---------------------------------------------------------------------------
// Elementwise kernels
// ---------------------------------------------------------------------------
__global__ void k_f32_to_bf16(const float* __restrict__ s, bf16* __restrict__ d, long long n) {
    long long i = (long long)blockIdx.x * blockDim.x + threadIdx.x;
    long long st = (long long)gridDim.x * blockDim.x;
    for (; i < n; i += st) d[i] = (bf16)s[i];
}

// dst[r][c] = src[(r/segSize)*segStride + segBase + r%segSize][c]   (bf16)
__global__ void k_gather_rows_bf16(const bf16* __restrict__ src, int srcLd,
                                   bf16* __restrict__ dst, int rows, int cols,
                                   int segSize, int segStride, int segBase) {
    long long n = (long long)rows * cols;
    long long i = (long long)blockIdx.x * blockDim.x + threadIdx.x;
    long long st = (long long)gridDim.x * blockDim.x;
    for (; i < n; i += st) {
        int rr = (int)(i / cols), c = (int)(i % cols);
        int g = (rr / segSize) * segStride + segBase + (rr % segSize);
        dst[i] = src[(long long)g * srcLd + c];
    }
}

// leaves: c = sig(i)*tanh(u); h = sig(o)*tanh(c)
__global__ void k_leaf(const float* __restrict__ XW, float* c_buf,
                       float* __restrict__ hf, bf16* __restrict__ hb) {
    long long n = (long long)T_TREES * LEAVES_PT * HDIM;
    long long i = (long long)blockIdx.x * blockDim.x + threadIdx.x;
    long long st = (long long)gridDim.x * blockDim.x;
    for (; i < n; i += st) {
        int c = (int)(i % HDIM);
        long long row = i / HDIM;
        int tree = (int)(row / LEAVES_PT), off = (int)(row % LEAVES_PT);
        long long g = (long long)tree * NPT + off;
        const float* xw = XW + g * H3;
        float iv = xw[c], ov = xw[HDIM + c], uv = xw[2 * HDIM + c];
        float cn = sigf(iv) * tanhf(uv);
        float hn = sigf(ov) * tanhf(cn);
        c_buf[g * HDIM + c] = cn;
        hf[g * HDIM + c] = hn;
        hb[g * HDIM + c] = (bf16)hn;
    }
}

// hsum_b[p][c] = bf16( sum_{k<4} h[f32] of children )   (children contiguous)
__global__ void k_hsum(const float* __restrict__ hf, bf16* __restrict__ hsum_b,
                       int cntP, int cStart) {
    long long n = (long long)T_TREES * cntP * HDIM;
    long long i = (long long)blockIdx.x * blockDim.x + threadIdx.x;
    long long st = (long long)gridDim.x * blockDim.x;
    for (; i < n; i += st) {
        int c = (int)(i % HDIM);
        long long p = i / HDIM;
        int tree = (int)(p / cntP), j = (int)(p % cntP);
        long long base = ((long long)tree * NPT + cStart + 4 * j) * HDIM + c;
        float s = hf[base] + hf[base + HDIM] + hf[base + 2 * HDIM] + hf[base + 3 * HDIM];
        hsum_b[i] = (bf16)s;
    }
}

// interior node: iou = XW[g] + HU[p]; f_k = sig(XFp[pp] + FH[cp_k]);
// c = sig(i)*tanh(u) + sum f_k*c_child ; h = sig(o)*tanh(c)
__global__ void k_level(const float* __restrict__ XW, const float* __restrict__ HU,
                        const float* __restrict__ XFp, const float* __restrict__ FH,
                        float* c_buf, float* __restrict__ hf, bf16* __restrict__ hb,
                        int cntP, int cntC, int pStart, int cStart, int pXFBase) {
    long long n = (long long)T_TREES * cntP * HDIM;
    long long i = (long long)blockIdx.x * blockDim.x + threadIdx.x;
    long long st = (long long)gridDim.x * blockDim.x;
    for (; i < n; i += st) {
        int c = (int)(i % HDIM);
        long long p = i / HDIM;
        int tree = (int)(p / cntP), j = (int)(p % cntP);
        long long g = (long long)tree * NPT + pStart + j;
        const float* xw = XW + g * H3;
        const float* hu = HU + p * H3;
        float iv = xw[c]            + hu[c];
        float ov = xw[HDIM + c]     + hu[HDIM + c];
        float uv = xw[2 * HDIM + c] + hu[2 * HDIM + c];
        long long pp = (long long)tree * PAR_PT + pXFBase + j;
        float xf = XFp[pp * HDIM + c];
        float csum = 0.0f;
        #pragma unroll
        for (int k = 0; k < 4; ++k) {
            long long cp = (long long)tree * cntC + 4 * j + k;
            long long gc = (long long)tree * NPT + cStart + 4 * j + k;
            float f = sigf(xf + FH[cp * HDIM + c]);
            csum += f * c_buf[gc * HDIM + c];
        }
        float cn = sigf(iv) * tanhf(uv) + csum;
        float hn = sigf(ov) * tanhf(cn);
        c_buf[g * HDIM + c] = cn;
        hf[g * HDIM + c] = hn;
        hb[g * HDIM + c] = (bf16)hn;
    }
}

// root head: softmax(h[root] @ W_ff^T), 64 trees x 4 classes, one block of 256
__global__ void k_root(const float* __restrict__ hf, const float* __restrict__ W_ff,
                       float* __restrict__ out_root) {
    __shared__ float lg[T_TREES][4];
    int t = threadIdx.x >> 2, k = threadIdx.x & 3;
    const float* hrow = hf + ((long long)t * NPT + (NPT - 1)) * HDIM;
    const float* w = W_ff + k * HDIM;
    float s = 0.0f;
    for (int c = 0; c < HDIM; ++c) s += hrow[c] * w[c];
    lg[t][k] = s;
    __syncthreads();
    float m = fmaxf(fmaxf(lg[t][0], lg[t][1]), fmaxf(lg[t][2], lg[t][3]));
    float d = expf(lg[t][0] - m) + expf(lg[t][1] - m) + expf(lg[t][2] - m) + expf(lg[t][3] - m);
    out_root[t * 4 + k] = expf(lg[t][k] - m) / d;
}

// stance head: logits = HST@W_sf^T (4), LN(4), softmax(4). One node per wave32.
__global__ __launch_bounds__(256)
void k_final(const float* __restrict__ HST, const float* __restrict__ W_sf,
             const float* __restrict__ gam, const float* __restrict__ bet,
             float* __restrict__ out) {
    int lane = threadIdx.x & 31;
    long long node = ((long long)blockIdx.x * blockDim.x + threadIdx.x) >> 5;
    if (node >= NTOT) return;
    const float* row = HST + node * HDIM;
    float s0 = 0.f, s1 = 0.f, s2 = 0.f, s3 = 0.f;
    for (int c = lane; c < HDIM; c += 32) {
        float x = row[c];
        s0 += x * W_sf[c];
        s1 += x * W_sf[HDIM + c];
        s2 += x * W_sf[2 * HDIM + c];
        s3 += x * W_sf[3 * HDIM + c];
    }
    #pragma unroll
    for (int m = 16; m >= 1; m >>= 1) {
        s0 += __shfl_xor(s0, m, 32);
        s1 += __shfl_xor(s1, m, 32);
        s2 += __shfl_xor(s2, m, 32);
        s3 += __shfl_xor(s3, m, 32);
    }
    if (lane == 0) {
        float mu = 0.25f * (s0 + s1 + s2 + s3);
        float d0 = s0 - mu, d1 = s1 - mu, d2 = s2 - mu, d3 = s3 - mu;
        float var = 0.25f * (d0 * d0 + d1 * d1 + d2 * d2 + d3 * d3);
        float rs = rsqrtf(var + 1e-6f);
        float y0 = d0 * rs * gam[0] + bet[0];
        float y1 = d1 * rs * gam[1] + bet[1];
        float y2 = d2 * rs * gam[2] + bet[2];
        float y3 = d3 * rs * gam[3] + bet[3];
        float m = fmaxf(fmaxf(y0, y1), fmaxf(y2, y3));
        float e0 = expf(y0 - m), e1 = expf(y1 - m), e2 = expf(y2 - m), e3 = expf(y3 - m);
        float inv = 1.0f / (e0 + e1 + e2 + e3);
        out[node * 4 + 0] = e0 * inv;
        out[node * 4 + 1] = e1 * inv;
        out[node * 4 + 2] = e2 * inv;
        out[node * 4 + 3] = e3 * inv;
    }
}

// ---------------------------------------------------------------------------
// Host orchestration
// ---------------------------------------------------------------------------
extern "C" void kernel_launch(void* const* d_in, const int* in_sizes, int n_in,
                              void* d_out, int out_size, void* d_ws, size_t ws_size,
                              hipStream_t stream) {
    (void)in_sizes; (void)n_in; (void)out_size; (void)ws_size;
    const float* features = (const float*)d_in[0];
    const float* W_iou = (const float*)d_in[6];
    const float* b_iou = (const float*)d_in[7];
    const float* U_iou = (const float*)d_in[8];
    const float* W_f   = (const float*)d_in[9];
    const float* b_f   = (const float*)d_in[10];
    const float* U_f   = (const float*)d_in[11];
    const float* W_ff  = (const float*)d_in[12];
    const float* W_sd  = (const float*)d_in[13];
    const float* W_sd2 = (const float*)d_in[14];
    const float* W_sf  = (const float*)d_in[15];
    const float* ln_g  = (const float*)d_in[16];
    const float* ln_b  = (const float*)d_in[17];

    float* out_hst  = (float*)d_out;                 // N x 4
    float* out_root = out_hst + (size_t)NTOT * 4;    // 64 x 4
    float* c_out    = out_root + (size_t)T_TREES * 4;// N x 512 (raw c output)

    // workspace arena
    size_t off = 0;
    char* wsb = (char*)d_ws;
    auto arena = [&](size_t bytes) -> void* {
        void* p = wsb + off;
        off += (bytes + 255) & ~(size_t)255;
        return p;
    };
    bf16* featb  = (bf16*)arena((size_t)NTOT * IN_FEAT * 2);
    bf16* Wiou_b = (bf16*)arena((size_t)H3 * IN_FEAT * 2);
    bf16* Uiou_b = (bf16*)arena((size_t)H3 * HDIM * 2);
    bf16* Wf_b   = (bf16*)arena((size_t)HDIM * IN_FEAT * 2);
    bf16* Uf_b   = (bf16*)arena((size_t)HDIM * HDIM * 2);
    bf16* Wsd_b  = (bf16*)arena((size_t)HS * HDIM * 2);
    bf16* Wsd2_b = (bf16*)arena((size_t)HDIM * HS * 2);
    bf16* pfeatb = (bf16*)arena((size_t)NPAR * IN_FEAT * 2);
    float* XW    = (float*)arena((size_t)NTOT * H3 * 4);
    float* XFp   = (float*)arena((size_t)NPAR * HDIM * 4);
    bf16*  hb    = (bf16*)arena((size_t)NTOT * HDIM * 2);
    float* hf    = (float*)arena((size_t)NTOT * HDIM * 4);
    bf16*  hsum_b  = (bf16*)arena((size_t)16384 * HDIM * 2);
    bf16*  childh_b= (bf16*)arena((size_t)65536 * HDIM * 2);
    float* HU    = (float*)arena((size_t)16384 * H3 * 4);
    float* FH    = (float*)arena((size_t)65536 * HDIM * 4);
    bf16*  S1b   = (bf16*)arena((size_t)NTOT * HS * 2);
    float* HST   = (float*)arena((size_t)NTOT * HDIM * 4);

    auto ewgrid = [](long long n) -> int {
        long long b = (n + 255) / 256;
        if (b > 262144) b = 262144;
        return (int)b;
    };
    auto conv = [&](const float* s, bf16* d, long long n) {
        k_f32_to_bf16<<<ewgrid(n), 256, 0, stream>>>(s, d, n);
    };
    auto gemm = [&](const bf16* A, int lda, const bf16* W, int ldw, const float* bias,
                    const float* Crsd, int ldr, float* Cf, int ldcf, bf16* Cb, int ldcb,
                    int M, int N, int K) {
        dim3 g((N + TN - 1) / TN, (M + TM - 1) / TM), b(256);
        gemm_bf16_wmma<<<g, b, 0, stream>>>(A, lda, W, ldw, bias, Crsd, ldr,
                                            Cf, ldcf, Cb, ldcb, M, N, K);
    };

    // 1) precision conversion
    conv(features, featb, (long long)NTOT * IN_FEAT);
    conv(W_iou, Wiou_b, (long long)H3 * IN_FEAT);
    conv(U_iou, Uiou_b, (long long)H3 * HDIM);
    conv(W_f,   Wf_b,   (long long)HDIM * IN_FEAT);
    conv(U_f,   Uf_b,   (long long)HDIM * HDIM);
    conv(W_sd,  Wsd_b,  (long long)HS * HDIM);
    conv(W_sd2, Wsd2_b, (long long)HDIM * HS);

    // 2) pack parent features (per-tree contiguous rows 1024..1364)
    k_gather_rows_bf16<<<ewgrid((long long)NPAR * IN_FEAT), 256, 0, stream>>>(
        featb, IN_FEAT, pfeatb, NPAR, IN_FEAT, PAR_PT, NPT, LEAVES_PT);

    // 3) shared projections: XW = X@W_iou^T + b (all nodes); XFp = Xpar@W_f^T + b_f
    gemm(featb, IN_FEAT, Wiou_b, IN_FEAT, b_iou, nullptr, 0, XW, H3, nullptr, 0,
         NTOT, H3, IN_FEAT);
    gemm(pfeatb, IN_FEAT, Wf_b, IN_FEAT, b_f, nullptr, 0, XFp, HDIM, nullptr, 0,
         NPAR, HDIM, IN_FEAT);

    // 4) level 0 (leaves)
    k_leaf<<<ewgrid((long long)T_TREES * LEAVES_PT * HDIM), 256, 0, stream>>>(
        XW, c_out, hf, hb);

    // 5) levels 1..5
    const int counts[6] = {1024, 256, 64, 16, 4, 1};
    const int starts[6] = {0, 1024, 1280, 1344, 1360, 1364};
    for (int n = 1; n <= 5; ++n) {
        int cntP = counts[n], cntC = counts[n - 1];
        int P = T_TREES * cntP, C = T_TREES * cntC;
        k_gather_rows_bf16<<<ewgrid((long long)C * HDIM), 256, 0, stream>>>(
            hb, HDIM, childh_b, C, HDIM, cntC, NPT, starts[n - 1]);
        k_hsum<<<ewgrid((long long)P * HDIM), 256, 0, stream>>>(
            hf, hsum_b, cntP, starts[n - 1]);
        gemm(hsum_b, HDIM, Uiou_b, HDIM, nullptr, nullptr, 0, HU, H3, nullptr, 0,
             P, H3, HDIM);
        gemm(childh_b, HDIM, Uf_b, HDIM, nullptr, nullptr, 0, FH, HDIM, nullptr, 0,
             C, HDIM, HDIM);
        k_level<<<ewgrid((long long)P * HDIM), 256, 0, stream>>>(
            XW, HU, XFp, FH, c_out, hf, hb,
            cntP, cntC, starts[n], starts[n - 1], starts[n] - LEAVES_PT);
    }

    // 6) heads
    k_root<<<1, 256, 0, stream>>>(hf, W_ff, out_root);
    gemm(hb, HDIM, Wsd_b, HDIM, nullptr, nullptr, 0, nullptr, 0, S1b, HS,
         NTOT, HS, HDIM);
    gemm(S1b, HS, Wsd2_b, HS, nullptr, hf, HDIM, HST, HDIM, nullptr, 0,
         NTOT, HDIM, HS);
    {
        long long waves = (long long)NTOT * 32;
        k_final<<<(int)((waves + 255) / 256), 256, 0, stream>>>(
            HST, W_sf, ln_g, ln_b, out_hst);
    }
}
